// MultiHeadAttention_69715909148834
// MI455X (gfx1250) — matrix operand order repaired and checked by reading, not compiled, not run on hardware
//
#include <hip/hip_runtime.h>

// ---------------------------------------------------------------------------
// MultiHeadAttention for MI455X (gfx1250): bf16 WMMA pipeline.
//   B=2, S=2048, D=1024, H=16, DV=64
//   scores scaled by sqrt(D)=32, masked_fill(value=1e-9) BEFORE softmax.
// ---------------------------------------------------------------------------

typedef __bf16 bf16;
typedef bf16  v16bf __attribute__((ext_vector_type(16)));
typedef bf16  v8bf  __attribute__((ext_vector_type(8)));
typedef float v8f   __attribute__((ext_vector_type(8)));

#define WMMA_BF16(a, b, c) \
  __builtin_amdgcn_wmma_f32_16x16x32_bf16(false, (a), false, (b), (short)0, (c), false, false)

static __device__ __forceinline__ bf16 f2bf(float x) {
  // round-to-nearest-even fp32 -> bf16
  unsigned u = __builtin_bit_cast(unsigned, x);
  unsigned r = u + 0x7FFFu + ((u >> 16) & 1u);
  unsigned short h = (unsigned short)(r >> 16);
  return __builtin_bit_cast(bf16, h);
}

// --------------------------- conversion kernels ----------------------------

__global__ void cvt_copy(const float* __restrict__ x, bf16* __restrict__ y, int n) {
  int i = blockIdx.x * blockDim.x + threadIdx.x;
  if (i < n) y[i] = f2bf(x[i]);
}

// W is [K=1024][N=1024] row-major fp32; write Wt[N][K] bf16 (so B-tiles are
// contiguous per output column).
__global__ void cvt_transpose(const float* __restrict__ W, bf16* __restrict__ Wt) {
  int i = blockIdx.x * blockDim.x + threadIdx.x; // over 1M
  int k = i >> 10, n = i & 1023;
  Wt[n * 1024 + k] = f2bf(W[i]);
}

// ------------------------------ GEMM kernel --------------------------------
// C[M=4096, N=1024] = A[M,1024]bf16 @ Wt^T + bias.
// Wave tile: 32 rows x 64 cols (8 WMMA accumulators), K chunked by 32.
// Ping-pong register double buffering, loop unrolled 2x so buffer indices are
// compile-time constants (no register-copy rotation, no WAR nop padding).
// mode 0: store bf16, head-split rows-major  [(b*16+h), s, dv]   (Q, K)
// mode 1: store bf16, head-split transposed  [(b*16+h), dv, s]   (V^T)
// mode 2: store fp32 row-major to d_out                           (final)
__global__ __launch_bounds__(256) void gemm_bias(
    const bf16* __restrict__ A, const bf16* __restrict__ Bt,
    const float* __restrict__ bias, bf16* __restrict__ outB,
    float* __restrict__ outF, int mode) {
  const int K = 1024;
  const int lane = threadIdx.x & 31;
  const int wave = threadIdx.x >> 5;
  const int wt   = blockIdx.x * 8 + wave;  // 2048 wave tiles
  const int tm   = wt >> 4;                // 0..127  (M/32)
  const int tn   = wt & 15;                // 0..15   (N/64)
  const int nl   = lane & 15;
  const int hi   = lane >> 4;              // lane half

  union uA { v16bf v; v8bf h[2]; };

  v8f c[2][4] = {};
  const bf16* arow0 = A + (size_t)(tm * 32 + nl) * K;
  const bf16* arow1 = A + (size_t)(tm * 32 + 16 + nl) * K;
  const bf16* brow  = Bt + (size_t)(tn * 64 + nl) * K + hi * 16;

  uA a0[2], a1[2];
  v16bf bb[2][4];

  auto loadChunk = [&](int buf, int kc) {
    a0[buf].h[0] = *(const v8bf*)(arow0 + kc + hi * 16);
    a0[buf].h[1] = *(const v8bf*)(arow0 + kc + 8 + hi * 16);
    a1[buf].h[0] = *(const v8bf*)(arow1 + kc + hi * 16);
    a1[buf].h[1] = *(const v8bf*)(arow1 + kc + 8 + hi * 16);
#pragma unroll
    for (int t = 0; t < 4; ++t)
      bb[buf][t] = *(const v16bf*)(brow + (size_t)t * 16 * K + kc);
  };
  auto compute = [&](int buf) {
#pragma unroll
    for (int t = 0; t < 4; ++t) {
      c[0][t] = WMMA_BF16(a0[buf].v, bb[buf][t], c[0][t]);
      c[1][t] = WMMA_BF16(a1[buf].v, bb[buf][t], c[1][t]);
    }
  };

  loadChunk(0, 0);
  for (int kc = 0; kc < K; kc += 64) {   // 16 iterations, 2 chunks each
    loadChunk(1, kc + 32);               // kc+32 <= K-32 always
    compute(0);
    if (kc + 64 < K) loadChunk(0, kc + 64);
    compute(1);
  }

#pragma unroll
  for (int t = 0; t < 4; ++t) {
    const int n  = tn * 64 + t * 16 + nl;
    const float bs = bias[n];
#pragma unroll
    for (int rb = 0; rb < 2; ++rb) {
#pragma unroll
      for (int r = 0; r < 8; ++r) {
        const int m = tm * 32 + rb * 16 + r + 8 * hi;
        const float val = c[rb][t][r] + bs;
        if (mode == 2) {
          outF[(size_t)m * 1024 + n] = val;
        } else {
          const int b_ = m >> 11, s = m & 2047;
          const int h = n >> 6, dv = n & 63;
          size_t idx;
          if (mode == 0) idx = ((size_t)(b_ * 16 + h) * 2048 + s) * 64 + dv;
          else           idx = ((size_t)(b_ * 16 + h) * 64 + dv) * 2048 + s;
          outB[idx] = f2bf(val);
        }
      }
    }
  }
}

// --------------------------- flash attention -------------------------------
// One wave owns 16 queries of one (b,h); streams all 2048 keys in chunks of
// 32.  K B-tiles are ping-pong double buffered (next chunk's tiles load while
// the softmax VALU section runs).  Online softmax with mask-fill value 1e-9.
__global__ __launch_bounds__(256) void attn_kernel(
    const bf16* __restrict__ Q, const bf16* __restrict__ Kp,
    const bf16* __restrict__ Vt, const unsigned char* __restrict__ mask,
    bf16* __restrict__ X) {
  __shared__ __align__(32) bf16 pbuf[8][16 * 32];  // per-wave P staging (8 KB)

  const int lane = threadIdx.x & 31;
  const int wave = threadIdx.x >> 5;
  const int wid  = blockIdx.x * 8 + wave;  // 4096 wave tiles
  const int bh   = wid >> 7;               // 0..31  (b*16+h)
  const int qt   = wid & 127;              // query tile within (b,h)
  const int nl   = lane & 15;
  const int hi   = lane >> 4;
  const int b_   = bh >> 4;
  const int h    = bh & 15;

  const unsigned char* mbase = mask + (size_t)b_ * 2048 * 2048;
  const bf16* kbase = Kp + (size_t)bh * 2048 * 64;
  const bf16* vbase = Vt + (size_t)bh * 64 * 2048;

  // Q A-tiles (16 queries x 64 dv), loaded once.
  const bf16* qrow = Q + ((size_t)bh * 2048 + qt * 16 + nl) * 64;
  union uA { v16bf v; v8bf h2[2]; };
  uA a0, a1;
  a0.h2[0] = *(const v8bf*)(qrow + hi * 16);
  a0.h2[1] = *(const v8bf*)(qrow + 8 + hi * 16);
  a1.h2[0] = *(const v8bf*)(qrow + 32 + hi * 16);
  a1.h2[1] = *(const v8bf*)(qrow + 40 + hi * 16);

  v8f o[4] = {};
  float mrun[8], lrun[8];
#pragma unroll
  for (int r = 0; r < 8; ++r) { mrun[r] = -1e30f; lrun[r] = 0.0f; }

  bf16* ph = pbuf[wave];

  v16bf kbuf[2][4];  // [buf][ b00, b10, b01, b11 ]
  auto loadK = [&](int buf, int kc) {
    const bf16* k0 = kbase + (size_t)(kc + nl) * 64;
    const bf16* k1 = kbase + (size_t)(kc + 16 + nl) * 64;
    kbuf[buf][0] = *(const v16bf*)(k0 + hi * 16);
    kbuf[buf][1] = *(const v16bf*)(k0 + 32 + hi * 16);
    kbuf[buf][2] = *(const v16bf*)(k1 + hi * 16);
    kbuf[buf][3] = *(const v16bf*)(k1 + 32 + hi * 16);
  };

  auto attn_step = [&](int kc, int cur, int nxt) {
    // ---- scores: S(16x32) = Q(16x64) @ K^T, two 16x16 C tiles ----
    v8f c0 = {}, c1 = {};
    c0 = WMMA_BF16(a0.v, kbuf[cur][0], c0);
    c0 = WMMA_BF16(a1.v, kbuf[cur][1], c0);
    c1 = WMMA_BF16(a0.v, kbuf[cur][2], c1);
    c1 = WMMA_BF16(a1.v, kbuf[cur][3], c1);

    // ---- issue next K chunk + this chunk's V tiles; softmax hides them ----
    if (kc + 32 < 2048) loadK(nxt, kc + 32);
    v16bf bv[4];
#pragma unroll
    for (int t = 0; t < 4; ++t)
      bv[t] = *(const v16bf*)(vbase + (size_t)(t * 16 + nl) * 2048 + kc + hi * 16);

    // ---- scale*32, masked_fill(1e-9), online softmax update ----
#pragma unroll
    for (int r = 0; r < 8; ++r) {
      const int row = qt * 16 + r + 8 * hi;  // query index in [0,2048)
      const unsigned char* mr = mbase + (size_t)row * 2048 + kc;
      float s0 = mr[nl]      ? c0[r] * 32.0f : 1e-9f;
      float s1 = mr[16 + nl] ? c1[r] * 32.0f : 1e-9f;

      float rm = fmaxf(s0, s1);
      rm = fmaxf(rm, __shfl_xor(rm, 1, 32));
      rm = fmaxf(rm, __shfl_xor(rm, 2, 32));
      rm = fmaxf(rm, __shfl_xor(rm, 4, 32));
      rm = fmaxf(rm, __shfl_xor(rm, 8, 32));

      const float mn = fmaxf(mrun[r], rm);
      const float al = __expf(mrun[r] - mn);
      mrun[r] = mn;
      s0 = __expf(s0 - mn);
      s1 = __expf(s1 - mn);

      float rs = s0 + s1;
      rs += __shfl_xor(rs, 1, 32);
      rs += __shfl_xor(rs, 2, 32);
      rs += __shfl_xor(rs, 4, 32);
      rs += __shfl_xor(rs, 8, 32);
      lrun[r] = lrun[r] * al + rs;

#pragma unroll
      for (int t = 0; t < 4; ++t) o[t][r] *= al;

      // C-layout -> LDS row-major P[16][32]
      ph[(r + 8 * hi) * 32 + nl]      = f2bf(s0);
      ph[(r + 8 * hi) * 32 + 16 + nl] = f2bf(s1);
    }

    asm volatile("s_wait_dscnt 0" ::: "memory");

    // ---- reload P as A-tile (16 queries x 32 keys) ----
    uA pa;
    pa.h2[0] = *(const v8bf*)(ph + nl * 32 + hi * 16);
    pa.h2[1] = *(const v8bf*)(ph + nl * 32 + 8 + hi * 16);

    // ---- O(16x64) += P(16x32) @ V(32x64); V stored transposed [dv][s] ----
#pragma unroll
    for (int t = 0; t < 4; ++t) o[t] = WMMA_BF16(pa.v, bv[t], o[t]);
  };

  loadK(0, 0);
  for (int kc = 0; kc < 2048; kc += 64) {  // 32 iterations, 2 chunks each
    attn_step(kc,      0, 1);
    attn_step(kc + 32, 1, 0);
  }

  // ---- epilogue: normalize, write X[(b*S+s)*D + h*64+dv] bf16 ----
#pragma unroll
  for (int t = 0; t < 4; ++t) {
#pragma unroll
    for (int r = 0; r < 8; ++r) {
      const int srow = qt * 16 + r + 8 * hi;
      const int col  = h * 64 + t * 16 + nl;
      const float v  = o[t][r] / lrun[r];
      X[((size_t)b_ * 2048 + srow) * 1024 + col] = f2bf(v);
    }
  }
}

// ------------------------------ launcher -----------------------------------

extern "C" void kernel_launch(void* const* d_in, const int* in_sizes, int n_in,
                              void* d_out, int out_size, void* d_ws, size_t ws_size,
                              hipStream_t stream) {
  const float* q  = (const float*)d_in[0];
  const float* k  = (const float*)d_in[1];
  const float* v  = (const float*)d_in[2];
  const unsigned char* xmask = (const unsigned char*)d_in[3];  // bool[B,S,S]
  const float* Wq = (const float*)d_in[4];
  const float* bq = (const float*)d_in[5];
  const float* Wk = (const float*)d_in[6];
  const float* bk = (const float*)d_in[7];
  const float* Wv = (const float*)d_in[8];
  const float* bv = (const float*)d_in[9];
  const float* Wf = (const float*)d_in[10];
  const float* bf_ = (const float*)d_in[11];

  char* ws = (char*)d_ws;
  const size_t MB = (size_t)1 << 20;
  bf16* qb  = (bf16*)(ws + 0 * MB);   // 8 MB each (4096x1024 bf16)
  bf16* kb  = (bf16*)(ws + 8 * MB);
  bf16* vb  = (bf16*)(ws + 16 * MB);
  bf16* WqT = (bf16*)(ws + 24 * MB);  // 2 MB each
  bf16* WkT = (bf16*)(ws + 26 * MB);
  bf16* WvT = (bf16*)(ws + 28 * MB);
  bf16* WfT = (bf16*)(ws + 30 * MB);
  bf16* Qp  = (bf16*)(ws + 32 * MB);  // [BH,S,DV]
  bf16* Kp  = (bf16*)(ws + 40 * MB);  // [BH,S,DV]
  bf16* Vt  = (bf16*)(ws + 48 * MB);  // [BH,DV,S]
  bf16* X   = (bf16*)(ws + 56 * MB);  // attn output [B*S, D]

  const int NE = 2 * 2048 * 1024;     // elements per q/k/v tensor
  cvt_copy<<<NE / 256, 256, 0, stream>>>(q, qb, NE);
  cvt_copy<<<NE / 256, 256, 0, stream>>>(k, kb, NE);
  cvt_copy<<<NE / 256, 256, 0, stream>>>(v, vb, NE);

  const int NW = 1024 * 1024;
  cvt_transpose<<<NW / 256, 256, 0, stream>>>(Wq, WqT);
  cvt_transpose<<<NW / 256, 256, 0, stream>>>(Wk, WkT);
  cvt_transpose<<<NW / 256, 256, 0, stream>>>(Wv, WvT);
  cvt_transpose<<<NW / 256, 256, 0, stream>>>(Wf, WfT);

  // projections: 4096x1024x1024 each, 32x64 wave tiles -> 256 blocks
  gemm_bias<<<256, 256, 0, stream>>>(qb, WqT, bq, Qp, nullptr, 0);
  gemm_bias<<<256, 256, 0, stream>>>(kb, WkT, bk, Kp, nullptr, 0);
  gemm_bias<<<256, 256, 0, stream>>>(vb, WvT, bv, Vt, nullptr, 1);

  // flash attention over all (b,h,query-tile)
  attn_kernel<<<512, 256, 0, stream>>>(Qp, Kp, Vt, xmask, X);

  // final projection, fp32 output + bias
  gemm_bias<<<256, 256, 0, stream>>>(X, WfT, bf_, nullptr, (float*)d_out, 2);
}
